// NeuralGDEForecaster_11622181503407
// MI455X (gfx1250) — matrix operand order repaired and checked by reading, not compiled, not run on hardware
//
#include <hip/hip_runtime.h>
#include <hip/hip_bf16.h>
#include <stdint.h>

// Problem constants (match reference)
#define NB   4
#define NNODE 5000
#define NT_  12
#define NFH  12
#define NH   128
#define NBN  20000   // NB*NNODE

typedef __attribute__((ext_vector_type(16))) __bf16 v16bf;
typedef __attribute__((ext_vector_type(8)))  float  v8f;

// ---------- helpers ----------
__device__ __forceinline__ unsigned short f2bf(float f) {
    union { float f; unsigned u; } c; c.f = f;
    unsigned u = c.u;
    return (unsigned short)((u + 0x7FFFu + ((u >> 16) & 1u)) >> 16);
}
__device__ __forceinline__ float wred(float v) {
#pragma unroll
    for (int o = 16; o > 0; o >>= 1) v += __shfl_xor(v, o, 32);
    return v;
}

// ---------- CSR build ----------
__global__ void zero_int_kernel(int* p, int n) {
    int i = blockIdx.x * blockDim.x + threadIdx.x;
    if (i < n) p[i] = 0;
}
__global__ void count_deg_kernel(const int* __restrict__ dst, int* __restrict__ cnt, int E) {
    int i = blockIdx.x * blockDim.x + threadIdx.x;
    if (i < E) atomicAdd(&cnt[dst[i]], 1);
}
__global__ void dinv_kernel(const int* __restrict__ cnt, float* __restrict__ dinv, int n) {
    int i = blockIdx.x * blockDim.x + threadIdx.x;
    if (i < n) dinv[i] = rsqrtf((float)cnt[i] + 1.0f);  // +1 = self loop
}
// single block, 1024 threads: exclusive scan of cnt -> rowptr, also fills cursor
__global__ void scan_kernel(const int* __restrict__ cnt, int* __restrict__ rowptr,
                            int* __restrict__ cursor, int n) {
    __shared__ int sm[1024];
    __shared__ int carry;
    if (threadIdx.x == 0) carry = 0;
    __syncthreads();
    for (int base = 0; base < n; base += 1024) {
        int i = base + (int)threadIdx.x;
        int v = (i < n) ? cnt[i] : 0;
        sm[threadIdx.x] = v;
        __syncthreads();
        for (int off = 1; off < 1024; off <<= 1) {
            int t = 0;
            if ((int)threadIdx.x >= off) t = sm[threadIdx.x - off];
            __syncthreads();
            sm[threadIdx.x] += t;
            __syncthreads();
        }
        int excl = sm[threadIdx.x] - v;
        if (i < n) { rowptr[i] = carry + excl; cursor[i] = carry + excl; }
        __syncthreads();
        if (threadIdx.x == 0) carry += sm[1023];
        __syncthreads();
    }
    if (threadIdx.x == 0) rowptr[n] = carry;
}
__global__ void fill_csr_kernel(const int* __restrict__ src, const int* __restrict__ dst,
                                int* __restrict__ cursor, int* __restrict__ csrc, int E) {
    int i = blockIdx.x * blockDim.x + threadIdx.x;
    if (i < E) {
        int pos = atomicAdd(&cursor[dst[i]], 1);
        csrc[pos] = src[i];
    }
}

// ---------- weight packing: f32 W(k,n) -> bf16 WMMA B-fragment order ----------
// layout: P[(((nt*KS)+kk)*32 + lane)*16 + i],  col = nt*16 + lane%16,
//         k = kk*32 + (lane/16)*16 + i     (B rows striped across lanes, 16 consecutive K per half-wave)
__global__ void pack_b_kernel(const float* __restrict__ W, unsigned short* __restrict__ P,
                              int K, int NOUT, int sK, int sN) {
    int idx = blockIdx.x * blockDim.x + threadIdx.x;
    int KS = K / 32;
    int total = (NOUT / 16) * KS * 512;
    if (idx >= total) return;
    int i    = idx & 15;
    int lane = (idx >> 4) & 31;
    int rest = idx >> 9;
    int kk = rest % KS;
    int nt = rest / KS;
    int col = nt * 16 + (lane & 15);
    int k   = kk * 32 + ((lane >> 4) << 4) + i;
    P[idx] = f2bf(W[(size_t)k * sK + (size_t)col * sN]);
}

// ---------- WMMA GEMM: Y[BN x NOUT] (f32) = Abf[BN x K] (bf16) @ Bpack ----------
template <int K, int NOUT>
__global__ __launch_bounds__(256) void gemm_wmma_kernel(const unsigned short* __restrict__ Abf,
                                                        const unsigned short* __restrict__ Bpack,
                                                        float* __restrict__ Y) {
    constexpr int KS = K / 32;
    constexpr int NTI = NOUT / 16;
    int wave = threadIdx.x >> 5;
    int lane = threadIdx.x & 31;
    int rb = blockIdx.x * 8 + wave;
    if (rb >= NBN / 16) return;         // wave-uniform exit: EXEC stays all-ones for WMMA
    int row0 = rb * 16;
    int r  = lane & 15;
    int ks = (lane >> 4) * 16;          // A: half-wave K split (K 0-15 vs 16-31)
    union AF { v16bf v; uint4 u[2]; };
    AF a[KS];
    const unsigned short* arow = Abf + (size_t)(row0 + r) * K;
#pragma unroll
    for (int kk = 0; kk < KS; ++kk) {
        a[kk].u[0] = *(const uint4*)(arow + kk * 32 + ks);
        a[kk].u[1] = *(const uint4*)(arow + kk * 32 + 8 + ks);
    }
    int rsel = (lane >> 4) * 8;
    int coll = lane & 15;
#pragma unroll
    for (int nt = 0; nt < NTI; ++nt) {
        v8f c = {};
#pragma unroll
        for (int kk = 0; kk < KS; ++kk) {
            AF b;
            const unsigned short* bp = Bpack + (((size_t)nt * KS + kk) * 32 + lane) * 16;
            b.u[0] = *(const uint4*)(bp);
            b.u[1] = *(const uint4*)(bp + 8);
            c = __builtin_amdgcn_wmma_f32_16x16x32_bf16(false, a[kk].v, false, b.v,
                                                        (short)0, c, false, false);
        }
        float* yb = Y + (size_t)(row0 + rsel) * NOUT + nt * 16 + coll;
#pragma unroll
        for (int j = 0; j < 8; ++j) yb[(size_t)j * NOUT] = c[j];
    }
}

// ---------- GCN aggregation over CSR (H=128), fused bias+activation+outputs ----------
// ACT: 1 = relu, 2 = tanh
template <int ACT>
__global__ __launch_bounds__(256) void agg_kernel(const float* __restrict__ Y,
                                                  const int* __restrict__ rowptr,
                                                  const int* __restrict__ csrc,
                                                  const float* __restrict__ dinv,
                                                  const float* __restrict__ bias,
                                                  float* __restrict__ outF,           // nullable
                                                  unsigned short* __restrict__ outB,  // nullable
                                                  float* __restrict__ acc,            // nullable
                                                  float accw, int accOv) {
    int wave = threadIdx.x >> 5, lane = threadIdx.x & 31;
    int node = blockIdx.x * 8 + wave;
    if (node >= NBN) return;
    float di = dinv[node];
    float4 v = ((const float4*)(Y + (size_t)node * NH))[lane];
    float sn = di * di;
    v.x *= sn; v.y *= sn; v.z *= sn; v.w *= sn;
    int e0 = rowptr[node], e1 = rowptr[node + 1];
    for (int e = e0; e < e1; ++e) {
        int s = csrc[e];
        float nw = di * dinv[s];
        float4 ys = ((const float4*)(Y + (size_t)s * NH))[lane];
        v.x += nw * ys.x; v.y += nw * ys.y; v.z += nw * ys.z; v.w += nw * ys.w;
    }
    float4 b = ((const float4*)bias)[lane];
    v.x += b.x; v.y += b.y; v.z += b.z; v.w += b.w;
    if (ACT == 1) { v.x = fmaxf(v.x, 0.f); v.y = fmaxf(v.y, 0.f); v.z = fmaxf(v.z, 0.f); v.w = fmaxf(v.w, 0.f); }
    else         { v.x = tanhf(v.x); v.y = tanhf(v.y); v.z = tanhf(v.z); v.w = tanhf(v.w); }
    size_t off = (size_t)node * NH + lane * 4;
    if (outF) *(float4*)(outF + off) = v;
    if (outB) {
        ushort4 u; u.x = f2bf(v.x); u.y = f2bf(v.y); u.z = f2bf(v.z); u.w = f2bf(v.w);
        *(ushort4*)(outB + off) = u;
    }
    if (acc) {
        float4* ap = (float4*)(acc + off);
        if (accOv) { float4 t; t.x = accw * v.x; t.y = accw * v.y; t.z = accw * v.z; t.w = accw * v.w; *ap = t; }
        else { float4 t = *ap; t.x += accw * v.x; t.y += accw * v.y; t.z += accw * v.z; t.w += accw * v.w; *ap = t; }
    }
}

// ---------- GCN layer 1: scalar aggregate + rank-1 expand (in-dim = 1, out = 64) ----------
__global__ __launch_bounds__(256) void gcn1_kernel(const float* __restrict__ x, int t,
                                                   const int* __restrict__ rowptr,
                                                   const int* __restrict__ csrc,
                                                   const float* __restrict__ dinv,
                                                   const float* __restrict__ w1,
                                                   const float* __restrict__ b1,
                                                   unsigned short* __restrict__ outB) {  // BN x 64
    int wave = threadIdx.x >> 5, lane = threadIdx.x & 31;
    int node = blockIdx.x * 8 + wave;
    if (node >= NBN) return;
    float di = dinv[node];
    int e0 = rowptr[node], e1 = rowptr[node + 1];
    float s = 0.f;
    for (int e = e0 + lane; e < e1; e += 32) {
        int sr = csrc[e];
        s += dinv[sr] * x[(size_t)sr * NT_ + t];
    }
    s = wred(s);
    s = di * s + di * di * x[(size_t)node * NT_ + t];
    ushort2 u;
    float h0 = fmaxf(s * w1[lane * 2 + 0] + b1[lane * 2 + 0], 0.f);
    float h1 = fmaxf(s * w1[lane * 2 + 1] + b1[lane * 2 + 1], 0.f);
    u.x = f2bf(h0); u.y = f2bf(h1);
    *(ushort2*)(outB + (size_t)node * 64 + lane * 2) = u;
}

// ---------- attention: init, per-step online-softmax update, finalize ----------
__global__ void att_init_kernel(float* __restrict__ m, float* __restrict__ d, float* __restrict__ S) {
    int i = blockIdx.x * blockDim.x + threadIdx.x;
    if (i < NBN * NH) S[i] = 0.f;
    if (i < NBN) { m[i] = -3.4e38f; d[i] = 0.f; }
}
__global__ __launch_bounds__(256) void att_update_kernel(const float* __restrict__ Yatt,
                                                         const float* __restrict__ ab1,
                                                         const float* __restrict__ aw2,
                                                         const float* __restrict__ ab2,
                                                         const float* __restrict__ h3,
                                                         float* __restrict__ m, float* __restrict__ d,
                                                         float* __restrict__ S) {
    int wave = threadIdx.x >> 5, lane = threadIdx.x & 31;
    int node = blockIdx.x * 8 + wave;
    if (node >= NBN) return;
    float4 y = ((const float4*)(Yatt + (size_t)node * NH))[lane];
    float4 b = ((const float4*)ab1)[lane];
    float4 w = ((const float4*)aw2)[lane];
    float dot = tanhf(y.x + b.x) * w.x + tanhf(y.y + b.y) * w.y +
                tanhf(y.z + b.z) * w.z + tanhf(y.w + b.w) * w.w;
    dot = wred(dot);
    float sc = dot + ab2[0];
    float mo = m[node];
    float mn = fmaxf(mo, sc);
    float alpha = __expf(mo - mn);
    float beta  = __expf(sc - mn);
    if (lane == 0) { m[node] = mn; d[node] = d[node] * alpha + beta; }
    size_t off = (size_t)node * NH + lane * 4;
    float4 h = *(const float4*)(h3 + off);
    float4* Sp = (float4*)(S + off);
    float4 sv = *Sp;
    sv.x = sv.x * alpha + beta * h.x; sv.y = sv.y * alpha + beta * h.y;
    sv.z = sv.z * alpha + beta * h.z; sv.w = sv.w * alpha + beta * h.w;
    *Sp = sv;
}
__global__ void att_final_kernel(const float* __restrict__ S, const float* __restrict__ d,
                                 unsigned short* __restrict__ ybf) {
    int i = blockIdx.x * blockDim.x + threadIdx.x;
    if (i < NBN * NH) ybf[i] = f2bf(S[i] / d[i >> 7]);
}

// ---------- GRU (h0 = 0) ----------
__global__ void gru_kernel(const float* __restrict__ Ygru, const float* __restrict__ bih,
                           const float* __restrict__ bhh, float* __restrict__ h,
                           unsigned short* __restrict__ ybf) {
    int i = blockIdx.x * blockDim.x + threadIdx.x;
    if (i >= NBN * NH) return;
    int node = i >> 7, j = i & 127;
    const float* yr = Ygru + (size_t)node * 384;
    float ir = yr[j]        + bih[j]        + bhh[j];
    float iz = yr[128 + j]  + bih[128 + j]  + bhh[128 + j];
    float in = yr[256 + j]  + bih[256 + j];
    float hn = bhh[256 + j];
    float r = 1.f / (1.f + __expf(-ir));
    float z = 1.f / (1.f + __expf(-iz));
    float nn = tanhf(in + r * hn);
    float hid = (1.f - z) * nn;     // + z*h0, h0 == 0
    h[i] = hid;
    ybf[i] = f2bf(hid);
}

// ---------- output head for a single time column ----------
__global__ __launch_bounds__(256) void out_head_kernel(const float* __restrict__ h,
                                                       const float* __restrict__ ow,
                                                       const float* __restrict__ ob,
                                                       float* __restrict__ out, int t) {
    int wave = threadIdx.x >> 5, lane = threadIdx.x & 31;
    int node = blockIdx.x * 8 + wave;
    if (node >= NBN) return;
    float4 hv = ((const float4*)(h + (size_t)node * NH))[lane];
    float4 w  = ((const float4*)ow)[lane];
    float dot = hv.x * w.x + hv.y * w.y + hv.z * w.z + hv.w * w.w;
    dot = wred(dot);
    if (lane == 0) out[(size_t)node * NFH + t] = dot + ob[0];
}

// ---------- RK4 stage-input prep: ybf = bf16(h + c*k) ----------
__global__ void prep_kernel(const float* __restrict__ h, const float* __restrict__ k,
                            float c, unsigned short* __restrict__ ybf) {
    int i = blockIdx.x * blockDim.x + threadIdx.x;
    if (i < NBN * NH) ybf[i] = f2bf(h[i] + c * k[i]);
}

// ---------- RK4 update fused with output head: h += c6*acc; out[:,t]; ybf = bf16(h) ----------
__global__ __launch_bounds__(256) void rk4_update_kernel(float* __restrict__ h,
                                                         const float* __restrict__ acc,
                                                         unsigned short* __restrict__ ybf,
                                                         const float* __restrict__ ow,
                                                         const float* __restrict__ ob,
                                                         float* __restrict__ out, int t, float c6) {
    int wave = threadIdx.x >> 5, lane = threadIdx.x & 31;
    int node = blockIdx.x * 8 + wave;
    if (node >= NBN) return;
    size_t off = (size_t)node * NH + lane * 4;
    float4 hv = *(float4*)(h + off);
    float4 av = *(const float4*)(acc + off);
    float4 w  = ((const float4*)ow)[lane];
    hv.x += c6 * av.x; hv.y += c6 * av.y; hv.z += c6 * av.z; hv.w += c6 * av.w;
    *(float4*)(h + off) = hv;
    ushort4 u; u.x = f2bf(hv.x); u.y = f2bf(hv.y); u.z = f2bf(hv.z); u.w = f2bf(hv.w);
    *(ushort4*)(ybf + off) = u;
    float dot = hv.x * w.x + hv.y * w.y + hv.z * w.z + hv.w * w.w;
    dot = wred(dot);
    if (lane == 0) out[(size_t)node * NFH + t] = dot + ob[0];
}

// =================================================================================
extern "C" void kernel_launch(void* const* d_in, const int* in_sizes, int n_in,
                              void* d_out, int out_size, void* d_ws, size_t ws_size,
                              hipStream_t stream) {
    (void)n_in; (void)out_size; (void)ws_size;
    const float* x       = (const float*)d_in[0];
    const float* gcn_w1  = (const float*)d_in[1];
    const float* gcn_b1  = (const float*)d_in[2];
    const float* gcn_w2  = (const float*)d_in[3];
    const float* gcn_b2  = (const float*)d_in[4];
    const float* gcn_w3  = (const float*)d_in[5];
    const float* gcn_b3  = (const float*)d_in[6];
    const float* att_w1  = (const float*)d_in[7];
    const float* att_b1  = (const float*)d_in[8];
    const float* att_w2  = (const float*)d_in[9];
    const float* att_b2  = (const float*)d_in[10];
    const float* gru_wih = (const float*)d_in[11];
    const float* gru_bih = (const float*)d_in[13];
    const float* gru_bhh = (const float*)d_in[14];
    const float* ode_w1  = (const float*)d_in[15];
    const float* ode_b1  = (const float*)d_in[16];
    const float* ode_w2  = (const float*)d_in[17];
    const float* ode_b2  = (const float*)d_in[18];
    const float* out_w   = (const float*)d_in[19];
    const float* out_b   = (const float*)d_in[20];
    const int*   ei      = (const int*)d_in[21];
    const int E = in_sizes[21] / 2;
    const int* esrc = ei;
    const int* edst = ei + E;
    float* out = (float*)d_out;

    // workspace carve-out
    char* p = (char*)d_ws;
    auto take = [&](size_t bytes) { char* r = p; p += (bytes + 255) & ~(size_t)255; return r; };
    int*   cnt    = (int*)take(sizeof(int) * NBN);
    int*   rowptr = (int*)take(sizeof(int) * (NBN + 1));
    int*   cursor = (int*)take(sizeof(int) * NBN);
    int*   csrc   = (int*)take(sizeof(int) * E);
    float* dinv   = (float*)take(sizeof(float) * NBN);
    unsigned short* ybf = (unsigned short*)take(sizeof(unsigned short) * NBN * NH);
    float* Y    = (float*)take(sizeof(float) * NBN * 384);
    float* h    = (float*)take(sizeof(float) * NBN * NH);
    float* kbuf = (float*)take(sizeof(float) * NBN * NH);
    float* acc  = (float*)take(sizeof(float) * NBN * NH);   // doubles as attention S
    float* mbuf = (float*)take(sizeof(float) * NBN);
    float* dbuf = (float*)take(sizeof(float) * NBN);
    unsigned short* pw2  = (unsigned short*)take(sizeof(unsigned short) * 64 * 128);
    unsigned short* pw3  = (unsigned short*)take(sizeof(unsigned short) * 128 * 128);
    unsigned short* patt = (unsigned short*)take(sizeof(unsigned short) * 128 * 128);
    unsigned short* pgru = (unsigned short*)take(sizeof(unsigned short) * 128 * 384);
    unsigned short* po1  = (unsigned short*)take(sizeof(unsigned short) * 128 * 128);
    unsigned short* po2  = (unsigned short*)take(sizeof(unsigned short) * 128 * 128);

    auto ew = [](int n) { return dim3((n + 255) / 256); };
    const dim3 blk(256);
    const dim3 wpn((NBN + 7) / 8);          // wave-per-node kernels
    const dim3 ggrid((NBN / 16 + 7) / 8);   // WMMA gemm: 8 waves/block, 16 rows/wave

    // --- CSR build + norms ---
    zero_int_kernel<<<ew(NBN), blk, 0, stream>>>(cnt, NBN);
    count_deg_kernel<<<ew(E), blk, 0, stream>>>(edst, cnt, E);
    dinv_kernel<<<ew(NBN), blk, 0, stream>>>(cnt, dinv, NBN);
    scan_kernel<<<1, 1024, 0, stream>>>(cnt, rowptr, cursor, NBN);
    fill_csr_kernel<<<ew(E), blk, 0, stream>>>(esrc, edst, cursor, csrc, E);

    // --- pack weights to bf16 WMMA B-fragments ---
    pack_b_kernel<<<ew(64 * 128), blk, 0, stream>>>(gcn_w2, pw2, 64, 128, 128, 1);
    pack_b_kernel<<<ew(128 * 128), blk, 0, stream>>>(gcn_w3, pw3, 128, 128, 128, 1);
    pack_b_kernel<<<ew(128 * 128), blk, 0, stream>>>(att_w1, patt, 128, 128, 128, 1);
    pack_b_kernel<<<ew(128 * 384), blk, 0, stream>>>(gru_wih, pgru, 128, 384, 1, 128); // W_ih^T
    pack_b_kernel<<<ew(128 * 128), blk, 0, stream>>>(ode_w1, po1, 128, 128, 128, 1);
    pack_b_kernel<<<ew(128 * 128), blk, 0, stream>>>(ode_w2, po2, 128, 128, 128, 1);

    // --- spatial encoder + online-softmax temporal attention ---
    att_init_kernel<<<ew(NBN * NH), blk, 0, stream>>>(mbuf, dbuf, acc);
    for (int t = 0; t < NT_; ++t) {
        gcn1_kernel<<<wpn, blk, 0, stream>>>(x, t, rowptr, csrc, dinv, gcn_w1, gcn_b1, ybf);
        gemm_wmma_kernel<64, 128><<<ggrid, blk, 0, stream>>>(ybf, pw2, Y);
        agg_kernel<1><<<wpn, blk, 0, stream>>>(Y, rowptr, csrc, dinv, gcn_b2,
                                               nullptr, ybf, nullptr, 0.f, 0);
        gemm_wmma_kernel<128, 128><<<ggrid, blk, 0, stream>>>(ybf, pw3, Y);
        agg_kernel<1><<<wpn, blk, 0, stream>>>(Y, rowptr, csrc, dinv, gcn_b3,
                                               kbuf, ybf, nullptr, 0.f, 0);
        gemm_wmma_kernel<128, 128><<<ggrid, blk, 0, stream>>>(ybf, patt, Y);
        att_update_kernel<<<wpn, blk, 0, stream>>>(Y, att_b1, att_w2, att_b2, kbuf,
                                                   mbuf, dbuf, acc);
    }
    att_final_kernel<<<ew(NBN * NH), blk, 0, stream>>>(acc, dbuf, ybf);

    // --- GRU (single step, h0 = 0) ---
    gemm_wmma_kernel<128, 384><<<ggrid, blk, 0, stream>>>(ybf, pgru, Y);
    gru_kernel<<<ew(NBN * NH), blk, 0, stream>>>(Y, gru_bih, gru_bhh, h, ybf);
    out_head_kernel<<<wpn, blk, 0, stream>>>(h, out_w, out_b, out, 0);

    // --- graph ODE, RK4 ---
    const float dt = (float)NFH / (float)(NFH - 1);
    const float cs[4] = {0.f, 0.5f * dt, 0.5f * dt, dt};
    const float ws[4] = {1.f, 2.f, 2.f, 1.f};
    for (int step = 0; step < NFH - 1; ++step) {
        for (int st = 0; st < 4; ++st) {
            if (st > 0)  // stage 0 input (bf16 of h) already in ybf
                prep_kernel<<<ew(NBN * NH), blk, 0, stream>>>(h, kbuf, cs[st], ybf);
            gemm_wmma_kernel<128, 128><<<ggrid, blk, 0, stream>>>(ybf, po1, Y);
            agg_kernel<2><<<wpn, blk, 0, stream>>>(Y, rowptr, csrc, dinv, ode_b1,
                                                   nullptr, ybf, nullptr, 0.f, 0);
            gemm_wmma_kernel<128, 128><<<ggrid, blk, 0, stream>>>(ybf, po2, Y);
            agg_kernel<2><<<wpn, blk, 0, stream>>>(Y, rowptr, csrc, dinv, ode_b2,
                                                   kbuf, nullptr, acc, ws[st], st == 0 ? 1 : 0);
        }
        rk4_update_kernel<<<wpn, blk, 0, stream>>>(h, acc, ybf, out_w, out_b, out,
                                                   step + 1, dt / 6.f);
    }
}
